// TransformerEncoderDecoderModel_13400297964038
// MI455X (gfx1250) — compile-verified, matrix operand order checked by reading, not run on hardware
//
#include <hip/hip_runtime.h>
#include <math.h>

// ---------------------------------------------------------------------------
// Model dims (fixed by the reference)
// ---------------------------------------------------------------------------
constexpr int Bn  = 8;            // batch
constexpr int Sn  = 512;          // seq len
constexpr int Cn  = 512;          // n_embd
constexpr int Hn  = 8;            // heads
constexpr int DHn = 64;           // head dim
constexpr int Ln  = 4;            // layers (enc and dec each)
constexpr int Vn  = 8192;         // vocab
constexpr int NR  = Bn * Sn;      // 4096 rows

// ---------------------------------------------------------------------------
// WMMA types (CDNA5 gfx1250, wave32): V_WMMA_F32_16X16X32_BF16
// ---------------------------------------------------------------------------
typedef __attribute__((ext_vector_type(16))) __bf16          v16bf;
typedef __attribute__((ext_vector_type(8)))  float           v8f;
typedef __attribute__((ext_vector_type(8)))  unsigned short  us8;   // 16 B
typedef __attribute__((ext_vector_type(4)))  unsigned short  us4;   // 8 B

union BF16x16 {
  unsigned short u[16];
  us8            h[2];     // two contiguous 8-element K-runs
  v16bf          v;
};

// native fptrunc f32->bf16 (RNE); gfx1250 has bf16 HW, clang lowers natively
__device__ __forceinline__ unsigned short f32_to_bf16(float f) {
  __bf16 b = (__bf16)f;
  return __builtin_bit_cast(unsigned short, b);
}

// convert 8 consecutive f32 (16B-aligned) -> us8 (one b128 store / vector reg)
__device__ __forceinline__ us8 cvt8v(const float* __restrict__ s) {
  const float4 f0 = ((const float4*)s)[0];
  const float4 f1 = ((const float4*)s)[1];
  us8 r;
  r[0] = f32_to_bf16(f0.x); r[1] = f32_to_bf16(f0.y);
  r[2] = f32_to_bf16(f0.z); r[3] = f32_to_bf16(f0.w);
  r[4] = f32_to_bf16(f1.x); r[5] = f32_to_bf16(f1.y);
  r[6] = f32_to_bf16(f1.z); r[7] = f32_to_bf16(f1.w);
  return r;
}

// ISA 7.12.2 16-bit fragment: lane-half h's 16 elements are K-runs
// [8h, 8h+7] and [16+8h, 16+8h+7] -> two ds_load_b128 from a K-contiguous row.
__device__ __forceinline__ void lds_frag(const unsigned short* __restrict__ row,
                                         int half, BF16x16& f) {
  f.h[0] = *(const us8*)(row + 8 * half);
  f.h[1] = *(const us8*)(row + 16 + 8 * half);
}

// same two runs, sourced from global f32 (d-contiguous) with convert
__device__ __forceinline__ void glb_frag(const float* __restrict__ base,
                                         int half, BF16x16& f) {
  f.h[0] = cvt8v(base + 8 * half);
  f.h[1] = cvt8v(base + 16 + 8 * half);
}

__device__ __forceinline__ v8f wmma_bf16(const BF16x16& a, const BF16x16& b, v8f c) {
  return __builtin_amdgcn_wmma_f32_16x16x32_bf16(
      false, a.v, false, b.v, (short)0, c, false, false);
}

// ---------------------------------------------------------------------------
// Kernel 1: fused linear-feature embedding
// ---------------------------------------------------------------------------
struct FeatArgs {
  const float* feat[5];
  const float* mask[4];
  const float* w[5];
  const float* bb[5];
  const float* g[5];
  const float* be[5];
  const float* embd;        // [5, C]
  float*       out;         // [NR, C]
};

__global__ __launch_bounds__(256) void feat_embed(FeatArgs A) {
  __shared__ float red[256], red2[256];
  __shared__ float s_mu, s_rv;
  const int row = blockIdx.x, tid = threadIdx.x;
  const int c0 = tid, c1 = tid + 256;
  float acc0 = 0.f, acc1 = 0.f;
  for (int f = 0; f < 5; ++f) {
    const float val = A.feat[f][row];
    const float e0 = val * A.w[f][c0] + A.bb[f][c0];
    const float e1 = val * A.w[f][c1] + A.bb[f][c1];
    red[tid] = e0 + e1;
    red2[tid] = e0 * e0 + e1 * e1;
    __syncthreads();
    for (int off = 128; off > 0; off >>= 1) {
      if (tid < off) { red[tid] += red[tid + off]; red2[tid] += red2[tid + off]; }
      __syncthreads();
    }
    if (tid == 0) {
      float mu = red[0] * (1.0f / Cn);
      float var = red2[0] * (1.0f / Cn) - mu * mu;
      s_mu = mu; s_rv = rsqrtf(var + 1e-5f);
    }
    __syncthreads();
    const float mu = s_mu, rv = s_rv;
    float o0 = (e0 - mu) * rv * A.g[f][c0] + A.be[f][c0] + A.embd[f * Cn + c0];
    float o1 = (e1 - mu) * rv * A.g[f][c1] + A.be[f][c1] + A.embd[f * Cn + c1];
    const float mk = (f < 4) ? A.mask[f][row] : 1.0f;
    acc0 += o0 * mk; acc1 += o1 * mk;
    __syncthreads();
  }
  A.out[row * Cn + c0] = acc0;
  A.out[row * Cn + c1] = acc1;
}

// ---------------------------------------------------------------------------
// Kernel 2: GEMM  Y[M,N] = act( X[M,K] @ W[K,N] + bias[N] )
// fp32 in/out, bf16 WMMA, f32 accumulate. Block tile 256x64, 8 waves;
// each wave owns 32 rows x 64 cols (8 accumulators), so the 4 B fragments
// are reused across 2 A fragments: 12 ds_load_b128 per 8 WMMAs.
// LDS: A row-major [m][k], B TRANSPOSED [n][k] (both K-contiguous fragments).
// M%256==0, N%64==0, K%32==0. act: 0=none, 1=exact GELU.
// ---------------------------------------------------------------------------
__global__ __launch_bounds__(256) void gemm_bias(
    const float* __restrict__ X, const float* __restrict__ W,
    const float* __restrict__ bias, float* __restrict__ Y,
    int M, int N, int K, int act) {
  constexpr int BM = 256, BN = 64, BK = 32;
  __shared__ alignas(16) unsigned short As [BM * BK];   // 16 KB, [m][k]
  __shared__ alignas(16) unsigned short Bst[BN * BK];   // 4 KB, [n][k] (transposed)
  const int tid = threadIdx.x;
  const int wave = tid >> 5, lane = tid & 31;
  const int half = lane >> 4, l16 = lane & 15;
  const int m0 = blockIdx.y * BM;
  const int n0 = blockIdx.x * BN;

  v8f acc[2][4];
  #pragma unroll
  for (int mt = 0; mt < 2; ++mt)
    #pragma unroll
    for (int nt = 0; nt < 4; ++nt)
      #pragma unroll
      for (int r = 0; r < 8; ++r) acc[mt][nt][r] = 0.f;

  for (int k0 = 0; k0 < K; k0 += BK) {
    // stage A: 256x32 elems as 1024 groups of 8; 4 groups/thread
    #pragma unroll
    for (int e = 0; e < 4; ++e) {
      int g = tid + 256 * e;
      int r = g >> 2, c8 = (g & 3) * 8;
      const float* src = &X[(size_t)(m0 + r) * K + k0 + c8];
      if (k0 + BK < K) __builtin_prefetch(src + BK, 0, 1);  // global_prefetch_b8
      *(us8*)&As[r * BK + c8] = cvt8v(src);                 // ds_store_b128
    }
    // stage B transposed: 32x64 elems as 512 float4 groups; 2/thread
    #pragma unroll
    for (int e = 0; e < 2; ++e) {
      int g = tid + 256 * e;
      int r = g >> 4, c4 = (g & 15) * 4;                    // r=k row, c4=n col
      float4 w4 = *(const float4*)&W[(size_t)(k0 + r) * N + n0 + c4];
      Bst[(c4 + 0) * BK + r] = f32_to_bf16(w4.x);
      Bst[(c4 + 1) * BK + r] = f32_to_bf16(w4.y);
      Bst[(c4 + 2) * BK + r] = f32_to_bf16(w4.z);
      Bst[(c4 + 3) * BK + r] = f32_to_bf16(w4.w);
    }
    __syncthreads();

    BF16x16 a[2];
    lds_frag(&As[(wave * 32 +  0 + l16) * BK], half, a[0]);
    lds_frag(&As[(wave * 32 + 16 + l16) * BK], half, a[1]);
    #pragma unroll
    for (int nt = 0; nt < 4; ++nt) {
      BF16x16 b;
      lds_frag(&Bst[(nt * 16 + l16) * BK], half, b);
      acc[0][nt] = wmma_bf16(a[0], b, acc[0][nt]);
      acc[1][nt] = wmma_bf16(a[1], b, acc[1][nt]);
    }
    __syncthreads();
  }

  // epilogue (C/D layout: lane<16 -> M=r, N=lane; lane>=16 -> M=r+8, N=lane-16)
  #pragma unroll
  for (int mt = 0; mt < 2; ++mt) {
    #pragma unroll
    for (int nt = 0; nt < 4; ++nt) {
      const int n = n0 + nt * 16 + l16;
      const float bv = bias ? bias[n] : 0.0f;
      #pragma unroll
      for (int r = 0; r < 8; ++r) {
        const int m = m0 + wave * 32 + mt * 16 + r + 8 * half;
        float v = acc[mt][nt][r] + bv;
        if (act) v = 0.5f * v * (1.0f + erff(v * 0.70710678118654752f));
        Y[(size_t)m * N + n] = v;
      }
    }
  }
}

// ---------------------------------------------------------------------------
// Kernel 3: multi-head attention, one (b,h) x 16-query tile per block.
// QK^T via WMMA (K frags straight from global, d-contiguous b128 loads),
// mask+softmax via shfl, P.V via WMMA with V staged transposed in LDS chunks.
// ---------------------------------------------------------------------------
__global__ __launch_bounds__(256) void attn_kernel(
    const float* __restrict__ Q, const float* __restrict__ Km,
    const float* __restrict__ Vm, const float* __restrict__ amask,
    float* __restrict__ O, int causal) {
  __shared__ alignas(16) float Ssc[16 * 512];           // 32 KB scores; later Vt+partials
  __shared__ alignas(16) unsigned short Ps[16 * 512];   // 16 KB P (bf16)
  __shared__ alignas(16) unsigned short Qs[16 * 64];    // 2 KB Q tile (bf16)
  __shared__ float invsum[16];

  const int tid = threadIdx.x;
  const int wave = tid >> 5, lane = tid & 31;
  const int half = lane >> 4, l16 = lane & 15;
  const int bh = blockIdx.y;
  const int b = bh / Hn, h = bh % Hn;
  const int q0 = blockIdx.x * 16;
  const int rowBase = b * Sn;
  const int hoff = h * DHn;

  // stage Q tile 16x64 (float4 read -> us4 store)
  {
    int idx = tid * 4;
    int m = idx >> 6, d = idx & 63;
    float4 q4 = *(const float4*)&Q[(size_t)(rowBase + q0 + m) * Cn + hoff + d];
    us4 v;
    v[0] = f32_to_bf16(q4.x); v[1] = f32_to_bf16(q4.y);
    v[2] = f32_to_bf16(q4.z); v[3] = f32_to_bf16(q4.w);
    *(us4*)&Qs[m * 64 + d] = v;
  }
  __syncthreads();

  // --- scores: wave w owns keys 64w..64w+63 ---
  {
    const int keyBase = wave * 64;
    v8f acc[4];
    #pragma unroll
    for (int nt = 0; nt < 4; ++nt)
      #pragma unroll
      for (int r = 0; r < 8; ++r) acc[nt][r] = 0.f;
    #pragma unroll
    for (int kk = 0; kk < DHn; kk += 32) {
      BF16x16 a;
      lds_frag(&Qs[l16 * 64 + kk], half, a);
      #pragma unroll
      for (int nt = 0; nt < 4; ++nt) {
        const int key = keyBase + nt * 16 + l16;
        BF16x16 bf;
        glb_frag(&Km[(size_t)(rowBase + key) * Cn + hoff + kk], half, bf);
        acc[nt] = wmma_bf16(a, bf, acc[nt]);
      }
    }
    #pragma unroll
    for (int nt = 0; nt < 4; ++nt) {
      const int n = keyBase + nt * 16 + l16;
      #pragma unroll
      for (int r = 0; r < 8; ++r) {
        const int m = r + 8 * half;
        Ssc[m * 512 + n] = acc[nt][r] * 0.125f;   // 1/sqrt(64)
      }
    }
  }
  __syncthreads();

  // --- mask + softmax: wave w -> rows 2w (lanes 0-15) and 2w+1 (lanes 16-31)
  {
    const int rrow = 2 * wave + half;
    float mx = -3.0e38f;
    for (int j = l16; j < Sn; j += 16) {
      float s = Ssc[rrow * 512 + j];
      if (causal && j > q0 + rrow) s = -1.0e30f;
      if (amask) s += (1.0f - amask[rowBase + j]) * -1.0e4f;
      Ssc[rrow * 512 + j] = s;
      mx = fmaxf(mx, s);
    }
    #pragma unroll
    for (int off = 1; off < 16; off <<= 1) mx = fmaxf(mx, __shfl_xor(mx, off, 32));
    float sum = 0.f;
    for (int j = l16; j < Sn; j += 16) {
      float e = __expf(Ssc[rrow * 512 + j] - mx);
      sum += e;
      Ps[rrow * 512 + j] = f32_to_bf16(e);
    }
    #pragma unroll
    for (int off = 1; off < 16; off <<= 1) sum += __shfl_xor(sum, off, 32);
    if (l16 == 0) invsum[rrow] = 1.0f / sum;
  }

  // --- P.V: V staged transposed Vt[d][k] in 128-key chunks (reuses Ssc) ---
  unsigned short* Vt = (unsigned short*)Ssc;   // 64x128 bf16 = 16 KB (first half)
  float* part = Ssc + 4096;                    // 2x16x64 f32 = 8 KB (second half)
  const int nt = wave & 3, kh = wave >> 2;
  const int d0 = nt * 16;
  v8f acc;
  #pragma unroll
  for (int r = 0; r < 8; ++r) acc[r] = 0.f;

  for (int c = 0; c < 4; ++c) {
    __syncthreads();   // softmax done (c==0) / previous chunk consumed
    // stage 128 keys x 64 dims: 2048 float4 groups, 8/thread; coalesced reads
    #pragma unroll
    for (int e = 0; e < 8; ++e) {
      int g = tid + 256 * e;
      int d4 = (g & 15) * 4, k = g >> 4;       // k in [0,128)
      float4 v4 = *(const float4*)&Vm[(size_t)(rowBase + c * 128 + k) * Cn + hoff + d4];
      Vt[(d4 + 0) * 128 + k] = f32_to_bf16(v4.x);
      Vt[(d4 + 1) * 128 + k] = f32_to_bf16(v4.y);
      Vt[(d4 + 2) * 128 + k] = f32_to_bf16(v4.z);
      Vt[(d4 + 3) * 128 + k] = f32_to_bf16(v4.w);
    }
    __syncthreads();
    #pragma unroll
    for (int ks = 0; ks < 64; ks += 32) {
      const int klocal = kh * 64 + ks;         // within chunk [0,128)
      const int kabs = c * 128 + klocal;
      BF16x16 a, bv;
      lds_frag(&Ps[l16 * 512 + kabs], half, a);
      lds_frag(&Vt[(d0 + l16) * 128 + klocal], half, bv);
      acc = wmma_bf16(a, bv, acc);
    }
  }

  // split-K partials into `part`, then combine
  #pragma unroll
  for (int r = 0; r < 8; ++r) {
    const int m = r + 8 * half;
    part[(kh * 16 + m) * 64 + d0 + l16] = acc[r];
  }
  __syncthreads();
  {
    int idx = tid * 4;                          // 16x64 f32, float4 per thread
    int m = idx >> 6, d = idx & 63;
    const float is = invsum[m];
    float4 o;
    o.x = (part[m * 64 + d + 0] + part[(16 + m) * 64 + d + 0]) * is;
    o.y = (part[m * 64 + d + 1] + part[(16 + m) * 64 + d + 1]) * is;
    o.z = (part[m * 64 + d + 2] + part[(16 + m) * 64 + d + 2]) * is;
    o.w = (part[m * 64 + d + 3] + part[(16 + m) * 64 + d + 3]) * is;
    *(float4*)&O[(size_t)(rowBase + q0 + m) * Cn + hoff + d] = o;
  }
}

// ---------------------------------------------------------------------------
// Kernel 4: y = LayerNorm(x [+ r]) * g + b ; one block per row; in-place safe
// ---------------------------------------------------------------------------
__global__ __launch_bounds__(256) void add_ln(
    const float* __restrict__ x, const float* __restrict__ r,
    const float* __restrict__ g, const float* __restrict__ b,
    float* __restrict__ y) {
  __shared__ float red[256], red2[256];
  __shared__ float s_mu, s_rv;
  const int row = blockIdx.x, tid = threadIdx.x;
  const int c0 = tid, c1 = tid + 256;
  float v0 = x[row * Cn + c0], v1 = x[row * Cn + c1];
  if (r) { v0 += r[row * Cn + c0]; v1 += r[row * Cn + c1]; }
  red[tid] = v0 + v1;
  red2[tid] = v0 * v0 + v1 * v1;
  __syncthreads();
  for (int off = 128; off > 0; off >>= 1) {
    if (tid < off) { red[tid] += red[tid + off]; red2[tid] += red2[tid + off]; }
    __syncthreads();
  }
  if (tid == 0) {
    float mu = red[0] * (1.0f / Cn);
    float var = red2[0] * (1.0f / Cn) - mu * mu;
    s_mu = mu; s_rv = rsqrtf(var + 1e-5f);
  }
  __syncthreads();
  const float mu = s_mu, rv = s_rv;
  y[row * Cn + c0] = (v0 - mu) * rv * g[c0] + b[c0];
  y[row * Cn + c1] = (v1 - mu) * rv * g[c1] + b[c1];
}

// ---------------------------------------------------------------------------
// Kernel 5: decoder token embedding d = wte[ids] + pos_emb[s]
// ---------------------------------------------------------------------------
__global__ __launch_bounds__(256) void embed_tokens(
    const int* __restrict__ ids, const float* __restrict__ wte,
    const float* __restrict__ pos, float* __restrict__ out) {
  const int row = blockIdx.x, tid = threadIdx.x;
  const int s = row % Sn;
  const int id = ids[row];
  out[row * Cn + tid]       = wte[(size_t)id * Cn + tid]       + pos[s * Cn + tid];
  out[row * Cn + tid + 256] = wte[(size_t)id * Cn + tid + 256] + pos[s * Cn + tid + 256];
}

// ---------------------------------------------------------------------------
// Host orchestration. Input order (setup_inputs dict order, params flattened
// depth-first in insertion order):
//   0..4  x,y,z,t,o [B*S]   5..8  mask_x..t   9 input_ids(i32)  10 attention_mask
//   11..30 lin_feats{x,y,z,t,o}: lin.w, lin.b, ln.g, ln.b
//   31 embd  32 wte  33 pos_emb
//   34+16l enc l: wq,bq,wk,bk,wv,bv,wp,bp, w1,b1,w2,b2, ln1g,ln1b,ln2g,ln2b
//   98+18l dec l: attn(8), mlp(4), ln1(2), ln2(2), ln3(2)
//   170,171 ln_f g,b   172 lm_head [C,V]
// ---------------------------------------------------------------------------
extern "C" void kernel_launch(void* const* d_in, const int* in_sizes, int n_in,
                              void* d_out, int out_size, void* d_ws, size_t ws_size,
                              hipStream_t stream) {
  (void)in_sizes; (void)n_in; (void)out_size; (void)ws_size;
  auto F = [&](int i) { return (const float*)d_in[i]; };

  const int* ids = (const int*)d_in[9];
  const float* amask = F(10);

  // workspace carve (floats): 7 * [4096,512] + [4096,2048]  ~= 88 MB
  float* p = (float*)d_ws;
  float* h   = p; p += (size_t)NR * Cn;        // encoder hidden
  float* dx  = p; p += (size_t)NR * Cn;        // decoder hidden
  float* qb  = p; p += (size_t)NR * Cn;
  float* kb  = p; p += (size_t)NR * Cn;
  float* vb  = p; p += (size_t)NR * Cn;
  float* ab  = p; p += (size_t)NR * Cn;        // attention out
  float* t1  = p; p += (size_t)NR * Cn;        // residual branch
  float* hid = p; p += (size_t)NR * 4 * Cn;    // MLP hidden

  auto gemm = [&](const float* X, const float* W, const float* bias, float* Y,
                  int M, int N, int K, int act) {
    dim3 grid(N / 64, M / 256);
    gemm_bias<<<grid, 256, 0, stream>>>(X, W, bias, Y, M, N, K, act);
  };
  auto ln = [&](const float* x, const float* r, const float* g, const float* b,
                float* y) {
    add_ln<<<NR, 256, 0, stream>>>(x, r, g, b, y);
  };
  auto attn = [&](const float* q, const float* k, const float* v,
                  const float* am, float* o, int causal) {
    dim3 grid(Sn / 16, Bn * Hn);
    attn_kernel<<<grid, 256, 0, stream>>>(q, k, v, am, o, causal);
  };

  // 1) fused feature embedding -> h
  {
    FeatArgs fa;
    for (int f = 0; f < 5; ++f) {
      fa.feat[f] = F(f);
      fa.w[f]  = F(11 + 4 * f + 0);
      fa.bb[f] = F(11 + 4 * f + 1);
      fa.g[f]  = F(11 + 4 * f + 2);
      fa.be[f] = F(11 + 4 * f + 3);
    }
    for (int f = 0; f < 4; ++f) fa.mask[f] = F(5 + f);
    fa.embd = F(31);
    fa.out = h;
    feat_embed<<<NR, 256, 0, stream>>>(fa);
  }

  // 2) encoder stack
  for (int l = 0; l < Ln; ++l) {
    const int a0 = 34 + 16 * l;
    gemm(h, F(a0 + 0), F(a0 + 1), qb, NR, Cn, Cn, 0);
    gemm(h, F(a0 + 2), F(a0 + 3), kb, NR, Cn, Cn, 0);
    gemm(h, F(a0 + 4), F(a0 + 5), vb, NR, Cn, Cn, 0);
    attn(qb, kb, vb, nullptr, ab, 0);
    gemm(ab, F(a0 + 6), F(a0 + 7), t1, NR, Cn, Cn, 0);
    ln(h, t1, F(a0 + 12), F(a0 + 13), h);
    gemm(h, F(a0 + 8), F(a0 + 9), hid, NR, 4 * Cn, Cn, 1);        // GELU
    gemm(hid, F(a0 + 10), F(a0 + 11), t1, NR, Cn, 4 * Cn, 0);
    ln(h, t1, F(a0 + 14), F(a0 + 15), h);
  }

  // 3) decoder embedding
  embed_tokens<<<NR, 256, 0, stream>>>(ids, F(32), F(33), dx);

  // 4) decoder stack (self- and cross-attention share the same attn weights)
  for (int l = 0; l < Ln; ++l) {
    const int a0 = 98 + 18 * l;
    // self-attention, causal + pad mask
    gemm(dx, F(a0 + 0), F(a0 + 1), qb, NR, Cn, Cn, 0);
    gemm(dx, F(a0 + 2), F(a0 + 3), kb, NR, Cn, Cn, 0);
    gemm(dx, F(a0 + 4), F(a0 + 5), vb, NR, Cn, Cn, 0);
    attn(qb, kb, vb, amask, ab, 1);
    gemm(ab, F(a0 + 6), F(a0 + 7), t1, NR, Cn, Cn, 0);
    ln(dx, t1, F(a0 + 12), F(a0 + 13), dx);
    // cross-attention vs encoder memory
    gemm(dx, F(a0 + 0), F(a0 + 1), qb, NR, Cn, Cn, 0);
    gemm(h,  F(a0 + 2), F(a0 + 3), kb, NR, Cn, Cn, 0);
    gemm(h,  F(a0 + 4), F(a0 + 5), vb, NR, Cn, Cn, 0);
    attn(qb, kb, vb, nullptr, ab, 0);
    gemm(ab, F(a0 + 6), F(a0 + 7), t1, NR, Cn, Cn, 0);
    ln(dx, t1, F(a0 + 14), F(a0 + 15), dx);
    // MLP
    gemm(dx, F(a0 + 8), F(a0 + 9), hid, NR, 4 * Cn, Cn, 1);       // GELU
    gemm(hid, F(a0 + 10), F(a0 + 11), t1, NR, Cn, 4 * Cn, 0);
    ln(dx, t1, F(a0 + 16), F(a0 + 17), dx);
  }

  // 5) final LN + lm_head -> d_out [B*S, V] fp32
  ln(dx, nullptr, F(170), F(171), dx);
  gemm(dx, F(172), nullptr, (float*)d_out, NR, Vn, Cn, 0);
}
